// MyBlock_55920474194294
// MI455X (gfx1250) — matrix operand order repaired and checked by reading, not compiled, not run on hardware
//
#include <hip/hip_runtime.h>

typedef __attribute__((ext_vector_type(16))) __bf16 v16bf;
typedef __attribute__((ext_vector_type(8)))  float  v8f;

static constexpr int Bn = 2, Cch = 64, Hh = 152, Ww = 272;
static constexpr int HW = Hh * Ww;
static constexpr int TW = 16, TH = 4;   // 4x16 = 64-pixel tile

__device__ __forceinline__ unsigned short f2bf(float f) {
  unsigned int u = __float_as_uint(f);
  unsigned int r = u + 0x7FFFu + ((u >> 16) & 1u);   // round-to-nearest-even
  return (unsigned short)(r >> 16);
}

__device__ __forceinline__ int iclamp(int v, int lo, int hi) {
  return v < lo ? lo : (v > hi ? hi : v);
}

// Convert conv weights [O][C][taps] (taps = ky*3+kx, =1 for 1x1) into bf16,
// laid out in exact WMMA 16x32 A-fragment order:
//   dst[((tap*MT + mtile)*2 + kstep)*512 + lane*16 + h]
// so each lane of a wave reads one contiguous, 32B-aligned v16bf per fragment
// (directly from global; these stay hot in WGP$/L2).
__global__ void swizzle_w(const float* __restrict__ src, unsigned short* __restrict__ dst,
                          int Mrows, int MT, int taps, int Csrc, int total) {
  int i = blockIdx.x * blockDim.x + threadIdx.x;
  if (i >= total) return;
  int h     = i & 15;
  int lane  = (i >> 4) & 31;
  int kstep = (i >> 9) & 1;
  int mtile = (i >> 10) % MT;
  int tap   = (i >> 10) / MT;
  int m     = lane & 15;
  int khigh = (lane >> 4) & 1;
  int k32   = (khigh << 4) | h;
  int o = mtile * 16 + m;
  int c = kstep * 32 + k32;
  float v = (o < Mrows) ? src[(o * Csrc + c) * taps + tap] : 0.f;
  dst[i] = f2bf(v);
}

template<bool RES>
__global__ __launch_bounds__(256) void dcn_stage(
    const float* __restrict__ in,        // stage input (x or t)
    const float* __restrict__ xres,      // original x (residual), stage2 only
    const unsigned short* __restrict__ wOffA,  // [9][2][2][512] bf16 A-frags
    const float* __restrict__ boff,            // 27
    const unsigned short* __restrict__ wA,     // [9][4][2][512] bf16 A-frags
    const float* __restrict__ bconv,           // 64
    const float* __restrict__ gg, const float* __restrict__ bb,
    const float* __restrict__ mmn, const float* __restrict__ vvn,
    const unsigned short* __restrict__ wrA,    // [4][2][512] bf16 A-frags (RES)
    const float* __restrict__ br,              // 64 (RES)
    float* __restrict__ out)
{
  // Halo of the input tile, channels innermost: a WMMA B-fragment for any tap
  // is 16 contiguous channels at one pixel -> one aligned 32B ds_load per lane.
  __shared__ alignas(32) unsigned short sH[6][32][64];   // 24.0 KB
  __shared__ alignas(32) unsigned short sS[2][64][64];   // 16.0 KB (double buffer)
  __shared__ float sOm[27][64];                          //  6.75 KB
  __shared__ float sScale[64], sShift[64];

  const int tid  = threadIdx.x;
  const int wave = tid >> 5;
  const int lane = tid & 31;
  const int x0 = blockIdx.x * TW;
  const int y0 = blockIdx.y * TH;
  const int b  = blockIdx.z;
  const float* inb = in + (size_t)b * Cch * HW;

  if (tid < 64) {
    float sc = gg[tid] * rsqrtf(vvn[tid] + 1e-5f);
    sScale[tid] = sc;
    sShift[tid] = bb[tid] - mmn[tid] * sc;
  }

  const int KY[9] = {0,0,0,1,1,1,2,2,2};
  const int KX[9] = {0,1,2,0,1,2,0,1,2};

  // ---- halo fill (once): rows y0-1..y0+4, cols x0-1..x0+16, branchless ----
  #pragma unroll
  for (int i = 0; i < 48; ++i) {             // 6*32*64 = 12288 = 48*256
    int e   = i * 256 + tid;
    int ch  = e / 192;                       // 0..63
    int pos = e - ch * 192;                  // 0..191
    int hr = pos >> 5, hc = pos & 31;        // row 0..5, col 0..31 (0..17 used)
    int yy = y0 + hr - 1;
    int xx = x0 + hc - 1;
    bool valid = (hc < 18) & (yy >= 0) & (yy < Hh) & (xx >= 0) & (xx < Ww);
    float v = inb[ch * HW + iclamp(yy, 0, Hh - 1) * Ww + iclamp(xx, 0, Ww - 1)];
    sH[hr][hc][ch] = f2bf(valid ? v : 0.f);
  }
  __syncthreads();

  // ---- Phase 1: offset conv (27(->32) x 576 GEMM) on WMMA, no more syncs ----
  const int mtO = wave >> 2, ntO = wave & 3;  // 8 C-tiles = 2 Mtiles x 4 Ntiles
  const int kk16 = (lane >> 4) << 4;          // channel sub-block per lane half
  v8f accO = {};
  #pragma unroll
  for (int tap = 0; tap < 9; ++tap) {
    const unsigned short* hp = &sH[ntO + KY[tap]][(lane & 15) + KX[tap]][kk16];
    v16bf b0 = *(const v16bf*)(hp);
    v16bf b1 = *(const v16bf*)(hp + 32);
    v16bf a0 = *(const v16bf*)(wOffA + ((tap * 2 + mtO) * 2 + 0) * 512 + lane * 16);
    v16bf a1 = *(const v16bf*)(wOffA + ((tap * 2 + mtO) * 2 + 1) * 512 + lane * 16);
    accO = __builtin_amdgcn_wmma_f32_16x16x32_bf16(false, a0, false, b0, (short)0, accO, false, false);
    accO = __builtin_amdgcn_wmma_f32_16x16x32_bf16(false, a1, false, b1, (short)0, accO, false, false);
  }
  {
    int col   = ntO * 16 + (lane & 15);
    int mbase = mtO * 16 + ((lane >> 4) << 3);
    #pragma unroll
    for (int r = 0; r < 8; ++r) {
      int row = mbase + r;
      if (row < 27) sOm[row][col] = accO[r] + boff[row];
    }
  }
  __syncthreads();

  // ---- Phase 2: deformable sampling + main 64x576 GEMM, 1 barrier/tap ----
  const int t0 = wave * 2, t1 = t0 + 1;       // 16 C-tiles, 2 per wave
  const int mt0 = t0 >> 2, nt0 = t0 & 3;
  const int mt1 = t1 >> 2, nt1 = t1 & 3;
  v8f acc0 = {}, acc1 = {};
  v8f accR0 = {}, accR1 = {};

  const int p  = tid >> 2;                    // this thread's pixel (0..63)
  const int c0 = (tid & 3) << 4;              // this thread's 16-channel group
  const float py = (float)(y0 + (p >> 4));
  const float px = (float)(x0 + (p & 15));

  for (int tap = 0; tap < 9; ++tap) {
    const int ky = KY[tap], kx = KX[tap];
    // per-pixel bilinear setup, in registers (4x redundant, no LDS/sync)
    float dy  = sOm[tap][p];
    float dx  = sOm[9 + tap][p];
    float msk = 1.f / (1.f + expf(-sOm[18 + tap][p]));
    float ys = py + (float)(ky - 1) + dy;
    float xs = px + (float)(kx - 1) + dx;
    float fy = floorf(ys), fx = floorf(xs);
    float ry = ys - fy, rx = xs - fx;
    int iy = (int)fy, ix = (int)fx;
    float wgt[4];
    const float* bp[4];
    #pragma unroll
    for (int cor = 0; cor < 4; ++cor) {
      int oy = cor >> 1, ox = cor & 1;
      int yc = iy + oy, xc = ix + ox;
      bool valid = (yc >= 0) & (yc <= Hh - 1) & (xc >= 0) & (xc <= Ww - 1);
      float w = (oy ? ry : 1.f - ry) * (ox ? rx : 1.f - rx) * msk;
      wgt[cor] = valid ? w : 0.f;
      bp[cor]  = inb + iclamp(yc, 0, Hh - 1) * Ww + iclamp(xc, 0, Ww - 1);
    }
    // gather-sample 16 channels, pack bf16, 2x ds_store_b128
    unsigned int pk[8];
    #pragma unroll
    for (int j = 0; j < 16; j += 2) {
      int ca = (c0 + j) * HW, cb = ca + HW;
      float va = wgt[0]*bp[0][ca] + wgt[1]*bp[1][ca] + wgt[2]*bp[2][ca] + wgt[3]*bp[3][ca];
      float vb = wgt[0]*bp[0][cb] + wgt[1]*bp[1][cb] + wgt[2]*bp[2][cb] + wgt[3]*bp[3][cb];
      pk[j >> 1] = (unsigned int)f2bf(va) | ((unsigned int)f2bf(vb) << 16);
    }
    uint4* dst = (uint4*)&sS[tap & 1][p][c0];
    dst[0] = make_uint4(pk[0], pk[1], pk[2], pk[3]);
    dst[1] = make_uint4(pk[4], pk[5], pk[6], pk[7]);
    __syncthreads();
    // WMMA: A-fragments straight from global (pre-swizzled), B from LDS
    const unsigned short* sSb = &sS[tap & 1][0][0];
    {
      v16bf a, bf;
      bf = *(const v16bf*)(sSb + (nt0 * 16 + (lane & 15)) * 64 + kk16);
      a  = *(const v16bf*)(wA + ((tap * 4 + mt0) * 2 + 0) * 512 + lane * 16);
      acc0 = __builtin_amdgcn_wmma_f32_16x16x32_bf16(false, a, false, bf, (short)0, acc0, false, false);
      bf = *(const v16bf*)(sSb + (nt0 * 16 + (lane & 15)) * 64 + 32 + kk16);
      a  = *(const v16bf*)(wA + ((tap * 4 + mt0) * 2 + 1) * 512 + lane * 16);
      acc0 = __builtin_amdgcn_wmma_f32_16x16x32_bf16(false, a, false, bf, (short)0, acc0, false, false);
      bf = *(const v16bf*)(sSb + (nt1 * 16 + (lane & 15)) * 64 + kk16);
      a  = *(const v16bf*)(wA + ((tap * 4 + mt1) * 2 + 0) * 512 + lane * 16);
      acc1 = __builtin_amdgcn_wmma_f32_16x16x32_bf16(false, a, false, bf, (short)0, acc1, false, false);
      bf = *(const v16bf*)(sSb + (nt1 * 16 + (lane & 15)) * 64 + 32 + kk16);
      a  = *(const v16bf*)(wA + ((tap * 4 + mt1) * 2 + 1) * 512 + lane * 16);
      acc1 = __builtin_amdgcn_wmma_f32_16x16x32_bf16(false, a, false, bf, (short)0, acc1, false, false);
    }
    // no trailing barrier: next tap writes the other sS buffer
  }

  if (RES) {  // fused residual 1x1 conv of original x (K=64 GEMM), buf 1
    const float* xb = xres + (size_t)b * Cch * HW
                    + (y0 + (p >> 4)) * Ww + (x0 + (p & 15));
    unsigned int pk[8];
    #pragma unroll
    for (int j = 0; j < 16; j += 2) {
      int ca = (c0 + j) * HW;
      pk[j >> 1] = (unsigned int)f2bf(xb[ca]) | ((unsigned int)f2bf(xb[ca + HW]) << 16);
    }
    uint4* dst = (uint4*)&sS[1][p][c0];
    dst[0] = make_uint4(pk[0], pk[1], pk[2], pk[3]);
    dst[1] = make_uint4(pk[4], pk[5], pk[6], pk[7]);
    __syncthreads();
    const unsigned short* sSb = &sS[1][0][0];
    v16bf a, bf;
    bf = *(const v16bf*)(sSb + (nt0 * 16 + (lane & 15)) * 64 + kk16);
    a  = *(const v16bf*)(wrA + (mt0 * 2 + 0) * 512 + lane * 16);
    accR0 = __builtin_amdgcn_wmma_f32_16x16x32_bf16(false, a, false, bf, (short)0, accR0, false, false);
    bf = *(const v16bf*)(sSb + (nt0 * 16 + (lane & 15)) * 64 + 32 + kk16);
    a  = *(const v16bf*)(wrA + (mt0 * 2 + 1) * 512 + lane * 16);
    accR0 = __builtin_amdgcn_wmma_f32_16x16x32_bf16(false, a, false, bf, (short)0, accR0, false, false);
    bf = *(const v16bf*)(sSb + (nt1 * 16 + (lane & 15)) * 64 + kk16);
    a  = *(const v16bf*)(wrA + (mt1 * 2 + 0) * 512 + lane * 16);
    accR1 = __builtin_amdgcn_wmma_f32_16x16x32_bf16(false, a, false, bf, (short)0, accR1, false, false);
    bf = *(const v16bf*)(sSb + (nt1 * 16 + (lane & 15)) * 64 + 32 + kk16);
    a  = *(const v16bf*)(wrA + (mt1 * 2 + 1) * 512 + lane * 16);
    accR1 = __builtin_amdgcn_wmma_f32_16x16x32_bf16(false, a, false, bf, (short)0, accR1, false, false);
  }

  // ------------------------- epilogue: BN+ReLU (+res) ----------------------
  const int nn    = lane & 15;
  const int mbase = (lane >> 4) << 3;
  {
    int col = nt0 * 16 + nn;
    int yy = y0 + (col >> 4), xx = x0 + (col & 15);
    #pragma unroll
    for (int r = 0; r < 8; ++r) {
      int o = mt0 * 16 + mbase + r;
      float v = acc0[r] + bconv[o];
      v = fmaxf(v * sScale[o] + sShift[o], 0.f);
      if (RES) v += accR0[r] + br[o];
      out[((size_t)b * Cch + o) * HW + yy * Ww + xx] = v;
    }
  }
  {
    int col = nt1 * 16 + nn;
    int yy = y0 + (col >> 4), xx = x0 + (col & 15);
    #pragma unroll
    for (int r = 0; r < 8; ++r) {
      int o = mt1 * 16 + mbase + r;
      float v = acc1[r] + bconv[o];
      v = fmaxf(v * sScale[o] + sShift[o], 0.f);
      if (RES) v += accR1[r] + br[o];
      out[((size_t)b * Cch + o) * HW + yy * Ww + xx] = v;
    }
  }
}

extern "C" void kernel_launch(void* const* d_in, const int* in_sizes, int n_in,
                              void* d_out, int out_size, void* d_ws, size_t ws_size,
                              hipStream_t stream) {
  (void)in_sizes; (void)n_in; (void)out_size; (void)ws_size;
  const float* x     = (const float*)d_in[0];
  const float* woff1 = (const float*)d_in[1];
  const float* boff1 = (const float*)d_in[2];
  const float* w1    = (const float*)d_in[3];
  const float* b1    = (const float*)d_in[4];
  const float* g1    = (const float*)d_in[5];
  const float* be1   = (const float*)d_in[6];
  const float* m1    = (const float*)d_in[7];
  const float* v1    = (const float*)d_in[8];
  const float* woff2 = (const float*)d_in[9];
  const float* boff2 = (const float*)d_in[10];
  const float* w2    = (const float*)d_in[11];
  const float* b2    = (const float*)d_in[12];
  const float* g2    = (const float*)d_in[13];
  const float* be2   = (const float*)d_in[14];
  const float* m2    = (const float*)d_in[15];
  const float* v2    = (const float*)d_in[16];
  const float* wr    = (const float*)d_in[17];
  const float* br    = (const float*)d_in[18];
  float* outp = (float*)d_out;

  // workspace: t buffer (fp32) + bf16 swizzled weights
  float* tbuf = (float*)d_ws;
  size_t tElems = (size_t)Bn * Cch * HW;
  unsigned short* wp = (unsigned short*)(tbuf + tElems);
  unsigned short* wOffA1 = wp; wp += 9 * 2 * 1024;
  unsigned short* wA1    = wp; wp += 9 * 4 * 1024;
  unsigned short* wOffA2 = wp; wp += 9 * 2 * 1024;
  unsigned short* wA2    = wp; wp += 9 * 4 * 1024;
  unsigned short* wrA    = wp; wp += 4 * 1024;

  swizzle_w<<<dim3(72),  dim3(256), 0, stream>>>(woff1, wOffA1, 27, 2, 9, Cch, 18432);
  swizzle_w<<<dim3(144), dim3(256), 0, stream>>>(w1,    wA1,    64, 4, 9, Cch, 36864);
  swizzle_w<<<dim3(72),  dim3(256), 0, stream>>>(woff2, wOffA2, 27, 2, 9, Cch, 18432);
  swizzle_w<<<dim3(144), dim3(256), 0, stream>>>(w2,    wA2,    64, 4, 9, Cch, 36864);
  swizzle_w<<<dim3(16),  dim3(256), 0, stream>>>(wr,    wrA,    64, 4, 1, Cch, 4096);

  dim3 grid(Ww / TW, Hh / TH, Bn);
  dcn_stage<false><<<grid, dim3(256), 0, stream>>>(
      x, nullptr, wOffA1, boff1, wA1, b1, g1, be1, m1, v1, nullptr, nullptr, tbuf);
  dcn_stage<true><<<grid, dim3(256), 0, stream>>>(
      tbuf, x, wOffA2, boff2, wA2, b2, g2, be2, m2, v2, wrA, br, outp);
}
